// ComplexAttentionLSNet_71975061946471
// MI455X (gfx1250) — matrix operand order repaired and checked by reading, not compiled, not run on hardware
//
#include <hip/hip_runtime.h>

#define NT 25
#define NXD 448
#define NYD 448
#define PXW 115
#define STRD 111
#define NPIX (NT*NXD*NYD)

#define AS1 __attribute__((address_space(1)))
#define AS3 __attribute__((address_space(3)))

typedef __attribute__((ext_vector_type(16))) __bf16 v16bf;
typedef __attribute__((ext_vector_type(8)))  float  v8f;
typedef unsigned int v4u __attribute__((ext_vector_type(4)));
typedef int v8i __attribute__((ext_vector_type(8)));
typedef int v4i __attribute__((ext_vector_type(4)));

union BF16x16 { v16bf v; unsigned short s[16]; };

__device__ __forceinline__ unsigned short f2bfbits(float f) {
  unsigned u = __builtin_bit_cast(unsigned, f);
  return (unsigned short)((u + 0x7FFFu + ((u >> 16) & 1u)) >> 16);
}

__device__ __forceinline__ v8f wmma_bf16(const BF16x16& a, const BF16x16& b, v8f c) {
  return __builtin_amdgcn_wmma_f32_16x16x32_bf16(false, a.v, false, b.v,
                                                 (short)0, c, false, false);
}

// async global -> LDS copy of one dword (per-lane, ASYNCcnt-tracked)
__device__ __forceinline__ void async_g2l_b32(const float* g, float* l) {
  __builtin_amdgcn_global_load_async_to_lds_b32((AS1 int*)g, (AS3 int*)l, 0, 0);
}

// TDM: load a [5 frames][5 rows][115 cols] f32 tile (strides 448^2 / 448 / 1)
// from global tensor into contiguous LDS at byte offset lds_off.
__device__ __forceinline__ void tdm_load_tile(const float* gsrc, unsigned lds_off) {
  unsigned long long ga = (unsigned long long)gsrc;
  v4u g0;
  g0[0] = 1u;                                           // count=1, user descriptor
  g0[1] = lds_off;                                      // lds_addr (bytes)
  g0[2] = (unsigned)(ga & 0xFFFFFFFFull);               // global_addr[31:0]
  g0[3] = (unsigned)((ga >> 32) & 0x1FFFFFFull) | (2u << 30);  // addr[56:32], type=2
  const unsigned td = 0x40000000u;                      // huge dims: OOB never fires
  const unsigned long long s0 = (unsigned long long)NYD;          // dim0 stride
  const unsigned long long s1 = (unsigned long long)NYD * NXD;    // dim1 stride
  v8i g1;
  g1[0] = (int)(2u << 16);                              // wg_mask=0, data_size=4B
  g1[1] = (int)((td & 0xFFFFu) << 16);                  // tensor_dim0 lo16
  g1[2] = (int)(((td >> 16) & 0xFFFFu) | ((td & 0xFFFFu) << 16));   // dim0 hi | dim1 lo
  g1[3] = (int)(((td >> 16) & 0xFFFFu) | (115u << 16)); // dim1 hi | tile_dim0=115
  g1[4] = (int)(5u | (5u << 16));                       // tile_dim1=5, tile_dim2=5
  g1[5] = (int)(s0 & 0xFFFFFFFFull);                    // dim0_stride lo32
  g1[6] = (int)(((s0 >> 32) & 0xFFFFu) | ((s1 & 0xFFFFu) << 16));   // s0 hi | s1 lo16
  g1[7] = (int)((s1 >> 16) & 0xFFFFFFFFull);            // s1[47:16]
  v4i g2;
  g2[0] = (int)td;                                      // tensor_dim2 (huge)
  g2[1] = 0;                                            // tensor_dim3
  g2[2] = 0;                                            // tensor_dim2_stride lo
  g2[3] = 0;                                            // stride hi | tile_dim3=0
  v4i g3; g3[0] = 0; g3[1] = 0; g3[2] = 0; g3[3] = 0;
  v8i g4; g4[0]=0; g4[1]=0; g4[2]=0; g4[3]=0; g4[4]=0; g4[5]=0; g4[6]=0; g4[7]=0;
  __builtin_amdgcn_tensor_load_to_lds(g0, g1, g2, g3, g4, 0);
}

// ---------------------------------------------------------------------------
// Kernel A: fused complex conv1(5x5) -> relu -> conv2(3x1x1) -> relu ->
//           conv3(1x1) -> p = x - den*relu(tau)/num_iter
// Tile: one frame t, 2 rows x, 64 cols y (128 pixels). 256 threads = 8 waves.
// ---------------------------------------------------------------------------
__global__ __launch_bounds__(256) void k_conv_p(
    const float* __restrict__ xr, const float* __restrict__ xi,
    const float* __restrict__ w1r, const float* __restrict__ w1i,
    const float* __restrict__ b1r, const float* __restrict__ b1i,
    const float* __restrict__ w2r, const float* __restrict__ w2i,
    const float* __restrict__ b2r, const float* __restrict__ b2i,
    const float* __restrict__ w3r, const float* __restrict__ w3i,
    const float* __restrict__ b3r, const float* __restrict__ b3i,
    const float* __restrict__ tau_w, const int* __restrict__ num_iter,
    float* __restrict__ p_out)
{
  __shared__ float s_xr[3][6][68];
  __shared__ float s_xi[3][6][68];
  __shared__ unsigned short s_xbr[3][6][68];          // bf16 bits of stage
  __shared__ unsigned short s_xbi[3][6][68];
  __shared__ unsigned short s_h1[3 * 2 * 16 * 128];   // bf16 bits
  __shared__ float s_h2[2 * 16 * 128];

  const int tid  = threadIdx.x;
  const int lane = tid & 31;
  const int wv   = tid >> 5;
  const int Mlane = lane & 15;
  const int khalf = lane >> 4;

  const int bid = blockIdx.x;
  const int t   = bid / (224 * 7);
  const int rem = bid % (224 * 7);
  const int x0  = (rem / 7) * 2;
  const int y0  = (rem % 7) * 64;

  // ---- stage x for 3 frames with +-2 halo via async global->LDS ----
  for (int idx = tid; idx < 3 * 6 * 68; idx += 256) {
    int f  = idx / 408;
    int rr = (idx % 408) / 68;
    int cc = idx % 68;
    int tf = t - 1 + f;
    int gx = x0 - 2 + rr;
    int gy = y0 - 2 + cc;
    if (tf >= 0 && tf < NT && gx >= 0 && gx < NXD && gy >= 0 && gy < NYD) {
      size_t g = ((size_t)tf * NXD + gx) * NYD + gy;
      async_g2l_b32(xr + g, &s_xr[f][rr][cc]);
      async_g2l_b32(xi + g, &s_xi[f][rr][cc]);
    } else {
      s_xr[f][rr][cc] = 0.f;
      s_xi[f][rr][cc] = 0.f;
    }
  }
  for (int i = tid; i < 3 * 2 * 16 * 128; i += 256) s_h1[i] = 0;

  // ---- precompute B fragments while async copies are in flight ----
  const int ccol = Mlane;   // output-channel column of this lane
  BF16x16 B1r[2], B1i[2];
#pragma unroll
  for (int kb = 0; kb < 2; ++kb) {
#pragma unroll
    for (int v = 0; v < 8; ++v) {
#pragma unroll
      for (int e = 0; e < 2; ++e) {
        int k = kb * 32 + (v >> 2) * 16 + khalf * 8 + (v & 3) * 2 + e;
        float fr = 0.f, fi = 0.f;
        if (ccol < 12) {
          if (k < 25)      { fr =  w1r[ccol * 25 + k];        fi = w1i[ccol * 25 + k]; }
          else if (k < 50) { int k2 = k - 25;
                             fr = -w1i[ccol * 25 + k2];       fi = w1r[ccol * 25 + k2]; }
        }
        B1r[kb].s[v * 2 + e] = f2bfbits(fr);
        B1i[kb].s[v * 2 + e] = f2bfbits(fi);
      }
    }
  }
  BF16x16 B2r[3], B2i[3];
#pragma unroll
  for (int kb = 0; kb < 3; ++kb) {
#pragma unroll
    for (int v = 0; v < 8; ++v) {
#pragma unroll
      for (int e = 0; e < 2; ++e) {
        int k = kb * 32 + (v >> 2) * 16 + khalf * 8 + (v & 3) * 2 + e;   // k < 96
        int part = k / 48, k2 = k % 48, dt = k2 >> 4, ci = k2 & 15;
        float fr = 0.f, fi = 0.f;
        if (ccol < 12 && ci < 12) {
          float wr = w2r[(ccol * 12 + ci) * 3 + dt];
          float wi = w2i[(ccol * 12 + ci) * 3 + dt];
          if (part == 0) { fr = wr;  fi = wi; }
          else           { fr = -wi; fi = wr; }
        }
        B2r[kb].s[v * 2 + e] = f2bfbits(fr);
        B2i[kb].s[v * 2 + e] = f2bfbits(fi);
      }
    }
  }
  const float b1rv = (ccol < 12) ? b1r[ccol] : 0.f;
  const float b1iv = (ccol < 12) ? b1i[ccol] : 0.f;
  const float b2rv = (ccol < 12) ? b2r[ccol] : 0.f;
  const float b2iv = (ccol < 12) ? b2i[ccol] : 0.f;

  __builtin_amdgcn_s_wait_asynccnt(0);
  __syncthreads();

  // ---- one-time f32 -> bf16 conversion of the stage ----
  for (int idx = tid; idx < 3 * 6 * 68; idx += 256) {
    int f  = idx / 408;
    int rr = (idx % 408) / 68;
    int cc = idx % 68;
    s_xbr[f][rr][cc] = f2bfbits(s_xr[f][rr][cc]);
    s_xbi[f][rr][cc] = f2bfbits(s_xi[f][rr][cc]);
  }
  __syncthreads();

  // ---- conv1 via WMMA: 24 M-tiles (3 frames x 128 px) ----
  for (int ti = wv; ti < 24; ti += 8) {
    int f  = ti >> 3;
    int tf = t - 1 + f;
    if (tf < 0 || tf >= NT) continue;        // wave-uniform; h1 stays zero
    int mt  = ti & 7;
    int pix = mt * 16 + Mlane;
    int lx  = pix >> 6, ly = pix & 63;

    BF16x16 A0, A1;
#pragma unroll
    for (int kb = 0; kb < 2; ++kb) {
      BF16x16& A = kb ? A1 : A0;
#pragma unroll
      for (int v = 0; v < 8; ++v) {
#pragma unroll
        for (int e = 0; e < 2; ++e) {
          int k = kb * 32 + (v >> 2) * 16 + khalf * 8 + (v & 3) * 2 + e;
          unsigned short bits = 0;
          if (k < 25)      bits = s_xbr[f][lx + k / 5][ly + k % 5];
          else if (k < 50) { int k2 = k - 25;
                             bits = s_xbi[f][lx + k2 / 5][ly + k2 % 5]; }
          A.s[v * 2 + e] = bits;
        }
      }
    }
    v8f ar = {}; v8f ai = {};
    ar = wmma_bf16(A0, B1r[0], ar);
    ar = wmma_bf16(A1, B1r[1], ar);
    ai = wmma_bf16(A0, B1i[0], ai);
    ai = wmma_bf16(A1, B1i[1], ai);
#pragma unroll
    for (int r2 = 0; r2 < 8; ++r2) {
      int px2 = mt * 16 + r2 + 8 * khalf;
      float vr = fmaxf(ar[r2] + b1rv, 0.f);
      float vi = fmaxf(ai[r2] + b1iv, 0.f);
      if (ccol >= 12) { vr = 0.f; vi = 0.f; }
      s_h1[((f * 2 + 0) * 16 + ccol) * 128 + px2] = f2bfbits(vr);
      s_h1[((f * 2 + 1) * 16 + ccol) * 128 + px2] = f2bfbits(vi);
    }
  }
  __syncthreads();

  // ---- conv2 via WMMA: 8 M-tiles, K = part(2) x dt(3) x ch(16) = 96 ----
  {
    int pixb = wv * 16 + Mlane;
    BF16x16 A[3];
#pragma unroll
    for (int kb = 0; kb < 3; ++kb) {
#pragma unroll
      for (int v = 0; v < 8; ++v) {
#pragma unroll
        for (int e = 0; e < 2; ++e) {
          int k = kb * 32 + (v >> 2) * 16 + khalf * 8 + (v & 3) * 2 + e;
          int part = k / 48, k2 = k % 48, dt = k2 >> 4, ci = k2 & 15;
          A[kb].s[v * 2 + e] = s_h1[((dt * 2 + part) * 16 + ci) * 128 + pixb];
        }
      }
    }
    v8f a2r = {}; v8f a2i = {};
#pragma unroll
    for (int kb = 0; kb < 3; ++kb) {
      a2r = wmma_bf16(A[kb], B2r[kb], a2r);
      a2i = wmma_bf16(A[kb], B2i[kb], a2i);
    }
#pragma unroll
    for (int r2 = 0; r2 < 8; ++r2) {
      int px2 = wv * 16 + r2 + 8 * khalf;
      float vr = fmaxf(a2r[r2] + b2rv, 0.f);
      float vi = fmaxf(a2i[r2] + b2iv, 0.f);
      if (ccol >= 12) { vr = 0.f; vi = 0.f; }
      s_h2[(0 * 16 + ccol) * 128 + px2] = vr;
      s_h2[(1 * 16 + ccol) * 128 + px2] = vi;
    }
  }
  __syncthreads();

  // ---- conv3 (1x1) + p epilogue: 128 pixels ----
  if (tid < 128) {
    float taufac = fmaxf(tau_w[0], 0.f) / (float)num_iter[0];
    int pix = tid, lx = pix >> 6, ly = pix & 63;
    float dr = b3r[0], di = b3i[0];
#pragma unroll
    for (int c = 0; c < 12; ++c) {
      float hr = s_h2[(0 * 16 + c) * 128 + pix];
      float hi = s_h2[(1 * 16 + c) * 128 + pix];
      dr += hr * w3r[c] - hi * w3i[c];
      di += hr * w3i[c] + hi * w3r[c];
    }
    float xrv = s_xr[1][lx + 2][ly + 2];
    float xiv = s_xi[1][lx + 2][ly + 2];
    size_t g = ((size_t)t * NXD + (x0 + lx)) * NYD + (y0 + ly);
    p_out[2 * g]     = xrv - dr * taufac;
    p_out[2 * g + 1] = xiv - di * taufac;
  }
}

// ---------------------------------------------------------------------------
// Kernel B: Gram matrices G[b] = A A^H for 80 complex 5x13225 patch matrices.
// Patch tiles are pulled into LDS with the Tensor Data Mover in 23 chunks of
// [5 frames][5 rows][115 cols]; always in-bounds so no edge handling needed.
// ---------------------------------------------------------------------------
__global__ __launch_bounds__(256) void k_gram(
    const float* __restrict__ xr, const float* __restrict__ xi,
    float* __restrict__ G)
{
  __shared__ float s_bufr[5][5][115];
  __shared__ float s_bufi[5][5][115];
  __shared__ float s_red[25 * 8];
  const int tid = threadIdx.x, lane = tid & 31, wv = tid >> 5;
  const int b  = blockIdx.x;          // 0..79
  const int tc = b / 16, pidx = b % 16;
  const int pi = pidx / 4, pj = pidx % 4;

  const unsigned ldr = (unsigned)(unsigned long long)(AS3 char*)&s_bufr[0][0][0];
  const unsigned ldi = (unsigned)(unsigned long long)(AS3 char*)&s_bufi[0][0][0];

  float acc[25];
#pragma unroll
  for (int e = 0; e < 25; ++e) acc[e] = 0.f;

  for (int ch = 0; ch < 23; ++ch) {
    __syncthreads();                               // LDS buffer free
    if (wv == 0) {
      size_t g = ((size_t)(tc * 5) * NXD + (pi * STRD + ch * 5)) * NYD + pj * STRD;
      tdm_load_tile(xr + g, ldr);
      tdm_load_tile(xi + g, ldi);
      __builtin_amdgcn_s_wait_tensorcnt(0);
    }
    __syncthreads();                               // tile visible to all waves
    for (int idx = tid; idx < 5 * 115; idx += 256) {
      int xl = idx / 115, y = idx % 115;
      float zr[5], zi[5];
#pragma unroll
      for (int r = 0; r < 5; ++r) {
        zr[r] = s_bufr[r][xl][y];
        zi[r] = s_bufi[r][xl][y];
      }
      int e = 0;
#pragma unroll
      for (int a = 0; a < 5; ++a)
#pragma unroll
        for (int b2 = a; b2 < 5; ++b2)
          acc[e++] += zr[a] * zr[b2] + zi[a] * zi[b2];
#pragma unroll
      for (int a = 0; a < 5; ++a)
#pragma unroll
        for (int b2 = a + 1; b2 < 5; ++b2)
          acc[e++] += zi[a] * zr[b2] - zr[a] * zi[b2];
    }
  }
#pragma unroll
  for (int e = 0; e < 25; ++e) {
    float v = acc[e];
    for (int off = 16; off > 0; off >>= 1) v += __shfl_down(v, off, 32);
    if (lane == 0) s_red[e * 8 + wv] = v;
  }
  __syncthreads();
  if (tid < 25) {
    float s = 0.f;
#pragma unroll
    for (int w = 0; w < 8; ++w) s += s_red[tid * 8 + w];
    G[b * 25 + tid] = s;
  }
}

// ---------------------------------------------------------------------------
// Kernel C: per-matrix eigendecomposition of the real 10x10 embedding of the
// 5x5 Hermitian Gram; build W = U diag(relu(s - relu(th)*smax)/s) U^H.
// ---------------------------------------------------------------------------
__global__ __launch_bounds__(128) void k_eig(
    const float* __restrict__ G, const float* __restrict__ thres,
    float* __restrict__ W)
{
  int b = threadIdx.x;
  if (b >= 80) return;

  float Gr[5][5], Gi[5][5];
  {
    int e = 0;
    for (int a = 0; a < 5; ++a)
      for (int b2 = a; b2 < 5; ++b2) { Gr[a][b2] = G[b * 25 + e]; Gr[b2][a] = Gr[a][b2]; ++e; }
    for (int a = 0; a < 5; ++a) Gi[a][a] = 0.f;
    for (int a = 0; a < 5; ++a)
      for (int b2 = a + 1; b2 < 5; ++b2) { Gi[a][b2] = G[b * 25 + e]; Gi[b2][a] = -Gi[a][b2]; ++e; }
  }
  float Am[10][10], Vm[10][10];
  for (int r = 0; r < 10; ++r)
    for (int c = 0; c < 10; ++c) {
      float v;
      if (r < 5 && c < 5)        v =  Gr[r][c];
      else if (r < 5)            v = -Gi[r][c - 5];
      else if (c < 5)            v =  Gi[r - 5][c];
      else                       v =  Gr[r - 5][c - 5];
      Am[r][c] = v;
      Vm[r][c] = (r == c) ? 1.f : 0.f;
    }
  // cyclic Jacobi
  for (int sweep = 0; sweep < 12; ++sweep) {
    for (int p = 0; p < 9; ++p) {
      for (int q = p + 1; q < 10; ++q) {
        float apq = Am[p][q];
        if (fabsf(apq) < 1e-20f) continue;
        float theta = (Am[q][q] - Am[p][p]) / (2.f * apq);
        float tt = 1.f / (fabsf(theta) + sqrtf(theta * theta + 1.f));
        if (theta < 0.f) tt = -tt;
        float cc = 1.f / sqrtf(tt * tt + 1.f);
        float ss = tt * cc;
        for (int k = 0; k < 10; ++k) {
          float akp = Am[k][p], akq = Am[k][q];
          Am[k][p] = cc * akp - ss * akq;
          Am[k][q] = ss * akp + cc * akq;
        }
        for (int k = 0; k < 10; ++k) {
          float apk = Am[p][k], aqk = Am[q][k];
          Am[p][k] = cc * apk - ss * aqk;
          Am[q][k] = ss * apk + cc * aqk;
        }
        for (int k = 0; k < 10; ++k) {
          float vkp = Vm[k][p], vkq = Vm[k][q];
          Vm[k][p] = cc * vkp - ss * vkq;
          Vm[k][q] = ss * vkp + cc * vkq;
        }
      }
    }
  }
  float sv[10], smax = 0.f;
  for (int k = 0; k < 10; ++k) { sv[k] = sqrtf(fmaxf(Am[k][k], 0.f)); smax = fmaxf(smax, sv[k]); }
  float th = fmaxf(thres[b], 0.f);
  float ratio[10];
  for (int k = 0; k < 10; ++k)
    ratio[k] = (sv[k] > 1e-30f) ? fmaxf(sv[k] - th * smax, 0.f) / sv[k] : 0.f;
  // Wr = E[0:5,0:5], Wi = E[5:10,0:5] of  E = V diag(ratio) V^T
  for (int r = 0; r < 5; ++r)
    for (int c = 0; c < 5; ++c) {
      float wr = 0.f, wi = 0.f;
      for (int k = 0; k < 10; ++k) {
        wr += ratio[k] * Vm[r][k] * Vm[c][k];
        wi += ratio[k] * Vm[r + 5][k] * Vm[c][k];
      }
      W[b * 50 + r * 5 + c]      = wr;
      W[b * 50 + 25 + r * 5 + c] = wi;
    }
}

// ---------------------------------------------------------------------------
// Kernel D: gather-fold q = W x, divide by overlap count, blend with p.
// ---------------------------------------------------------------------------
__global__ __launch_bounds__(256) void k_final(
    const float* __restrict__ xr, const float* __restrict__ xi,
    const float* __restrict__ p_ws, const float* __restrict__ W,
    const float* __restrict__ p_w, float* __restrict__ out)
{
  size_t gid = (size_t)blockIdx.x * 256 + threadIdx.x;
  if (gid >= (size_t)NPIX) return;
  int t = (int)(gid / (NXD * NYD));
  int r2 = (int)(gid % (NXD * NYD));
  int X = r2 / NYD, Y = r2 % NYD;
  int tc = t / 5, rr = t % 5;

  float zr[5], zi[5];
#pragma unroll
  for (int rp = 0; rp < 5; ++rp) {
    size_t g = ((size_t)(tc * 5 + rp) * NXD + X) * NYD + Y;
    zr[rp] = xr[g]; zi[rp] = xi[g];
  }
  float qR = 0.f, qI = 0.f; int cnt = 0;
#pragma unroll
  for (int i = 0; i < 4; ++i) {
    int dx = X - i * STRD;
    if (dx < 0 || dx >= PXW) continue;
#pragma unroll
    for (int j = 0; j < 4; ++j) {
      int dy = Y - j * STRD;
      if (dy < 0 || dy >= PXW) continue;
      int b = tc * 16 + i * 4 + j;
      const float* Wb = W + b * 50;
#pragma unroll
      for (int rp = 0; rp < 5; ++rp) {
        float wr = Wb[rr * 5 + rp];
        float wi = Wb[25 + rr * 5 + rp];
        qR += wr * zr[rp] - wi * zi[rp];
        qI += wr * zi[rp] + wi * zr[rp];
      }
      ++cnt;
    }
  }
  float inv = 1.f / (float)cnt;
  qR *= inv; qI *= inv;
  float pw = p_w[0], rpw = fmaxf(pw, 0.f), qw = 1.f - pw;
  float pR = p_ws[2 * gid], pI = p_ws[2 * gid + 1];
  out[2 * gid]     = pR * rpw + qR * qw;
  out[2 * gid + 1] = pI * rpw + qI * qw;
}

// ---------------------------------------------------------------------------
extern "C" void kernel_launch(void* const* d_in, const int* in_sizes, int n_in,
                              void* d_out, int out_size, void* d_ws, size_t ws_size,
                              hipStream_t stream) {
  const float* xr  = (const float*)d_in[0];
  const float* xi  = (const float*)d_in[1];
  const float* w1r = (const float*)d_in[2];
  const float* w1i = (const float*)d_in[3];
  const float* b1r = (const float*)d_in[4];
  const float* b1i = (const float*)d_in[5];
  const float* w2r = (const float*)d_in[6];
  const float* w2i = (const float*)d_in[7];
  const float* b2r = (const float*)d_in[8];
  const float* b2i = (const float*)d_in[9];
  const float* w3r = (const float*)d_in[10];
  const float* w3i = (const float*)d_in[11];
  const float* b3r = (const float*)d_in[12];
  const float* b3i = (const float*)d_in[13];
  const float* thres = (const float*)d_in[14];
  const float* tau_w = (const float*)d_in[15];
  const float* p_w   = (const float*)d_in[16];
  const int*   niter = (const int*)d_in[17];

  float* ws  = (float*)d_ws;
  float* p_s = ws;                               // 2*NPIX floats
  float* G   = ws + (size_t)2 * NPIX;            // 80*25
  float* W   = G + 80 * 25;                      // 80*50

  k_conv_p<<<dim3(25 * 224 * 7), 256, 0, stream>>>(
      xr, xi, w1r, w1i, b1r, b1i, w2r, w2i, b2r, b2i,
      w3r, w3i, b3r, b3i, tau_w, niter, p_s);
  k_gram<<<dim3(80), 256, 0, stream>>>(xr, xi, G);
  k_eig<<<dim3(1), 128, 0, stream>>>(G, thres, W);
  k_final<<<dim3(NPIX / 256), 256, 0, stream>>>(xr, xi, p_s, W, p_w, (float*)d_out);
}
